// Sequence_55482387530371
// MI455X (gfx1250) — matrix operand order repaired and checked by reading, not compiled, not run on hardware
//
#include <hip/hip_runtime.h>

#define HID   32
#define BROWS 16     // batch rows per block (one WMMA M tile)
#define GS    132    // gate LDS row stride in floats (128 + pad -> conflict-free)
#define HSTR  40     // h LDS row stride in halves (32 + pad, keeps 16B alignment)

typedef __attribute__((ext_vector_type(16))) _Float16 v16h;
typedef __attribute__((ext_vector_type(8)))  _Float16 v8h;
typedef __attribute__((ext_vector_type(8)))  float    v8f;

#if __has_builtin(__builtin_amdgcn_tanhf)
// CDNA5 native v_tanh_f32: single-transcendental tanh and sigmoid
__device__ __forceinline__ float ftanh(float x) { return __builtin_amdgcn_tanhf(x); }
__device__ __forceinline__ float fsig(float x)  { return __builtin_fmaf(0.5f, __builtin_amdgcn_tanhf(0.5f * x), 0.5f); }
#else
__device__ __forceinline__ float fexp(float x)  { return __builtin_amdgcn_exp2f(x * 1.44269504088896340736f); }
__device__ __forceinline__ float fsig(float x)  { return __builtin_amdgcn_rcpf(1.0f + fexp(-x)); }
__device__ __forceinline__ float ftanh(float x) { return 2.0f * fsig(2.0f * x) - 1.0f; }
#endif

__global__ __launch_bounds__(256) void lstm_seq_kernel(
    const float* __restrict__ input,   // [B, T] (trailing dim 1 squeezed)
    const float* __restrict__ w_ih,    // [128]  (input size 1)
    const float* __restrict__ w_hh,    // [128, 32]
    const float* __restrict__ b_ih,    // [128]
    const float* __restrict__ b_hh,    // [128]
    const float* __restrict__ w_lin,   // [32]
    const float* __restrict__ b_lin,   // [1]
    float* __restrict__ out,           // [B, T+future]
    int T, int future)
{
    __shared__ float                  G[BROWS * GS];     // gate pre-activations [16 x 128 padded]
    __shared__ __align__(16) _Float16 Hs[BROWS * HSTR];  // h state (f16, WMMA-A friendly)
    __shared__ __align__(16) float    Xs[BROWS];         // current scalar input per batch row

    const int tid   = threadIdx.x;
    const int wv    = tid >> 5;          // wave id 0..7  -> gate column tile
    const int lane  = tid & 31;
    const int half  = lane >> 4;         // lane half (WMMA layout)
    const int nl    = lane & 15;
    const int bbase = blockIdx.x * BROWS;
    const int TT    = T + future;

    // ---------------- per-lane constants ----------------
    const int   col   = wv * 16 + nl;            // gate column 0..127 (i|f|g|o)
    const float wihc  = w_ih[col];
    const float biasc = b_ih[col] + b_hh[col];
    const float wlinj = w_lin[lane];
    const float blin  = b_lin[0];

    // B fragment = w_hh^T in WMMA B layout (32x16 f16): lane n = col,
    // element e -> K = e + 8*half + (e & 8)  (half0: K=0..7,16..23  half1: K=8..15,24..31)
    v16h bfrag;
    #pragma unroll
    for (int e = 0; e < 16; ++e)
        bfrag[e] = (_Float16)w_hh[col * HID + (e + 8 * half + (e & 8))];

    // ---------------- state init ----------------
    float c0 = 0.0f, c1 = 0.0f;   // cell state for rows (wv) and (wv+8); lane = hidden index
    for (int i = tid; i < BROWS * HSTR; i += 256) Hs[i] = (_Float16)0.0f;
    if (tid < BROWS) Xs[tid] = input[(size_t)(bbase + tid) * T];
    __syncthreads();

    for (int t = 0; t < TT; ++t) {
        // ---- accumulator init: x * w_ih + (b_ih + b_hh); row M = v + 8*half ----
        v8f acc;
        const float* xh = &Xs[8 * half];
        #pragma unroll
        for (int v = 0; v < 8; ++v)
            acc[v] = xh[v] * wihc + biasc;

        // ---- A fragment: h rows M = nl; half0: K=0..7,16..23  half1: K=8..15,24..31 ----
        const _Float16* hrow = &Hs[nl * HSTR + 8 * half];
        v8h alo = *(const v8h*)hrow;          // 16B aligned
        v8h ahi = *(const v8h*)(hrow + 16);   // +32B
        v16h afrag = __builtin_shufflevector(alo, ahi,
            0, 1, 2, 3, 4, 5, 6, 7, 8, 9, 10, 11, 12, 13, 14, 15);

        // ---- gates[16,16] tile = h @ w_hh^T + C  (fp32 accumulate) ----
        acc = __builtin_amdgcn_wmma_f32_16x16x32_f16(
            false, afrag, false, bfrag, (short)0, acc, false, false);

        // scatter gate tile: D layout -> row M = v + 8*half, col = col
        #pragma unroll
        for (int v = 0; v < 8; ++v)
            G[(v + 8 * half) * GS + col] = acc[v];

        __syncthreads();

        // ---- elementwise LSTM update: wave wv owns rows {wv, wv+8}, lane = hidden j ----
        const int r0 = wv, r1 = wv + 8;
        float i0 = G[r0 * GS + lane], f0 = G[r0 * GS + 32 + lane];
        float g0 = G[r0 * GS + 64 + lane], o0 = G[r0 * GS + 96 + lane];
        float i1 = G[r1 * GS + lane], f1 = G[r1 * GS + 32 + lane];
        float g1 = G[r1 * GS + 64 + lane], o1 = G[r1 * GS + 96 + lane];

        c0 = fsig(f0) * c0 + fsig(i0) * ftanh(g0);
        c1 = fsig(f1) * c1 + fsig(i1) * ftanh(g1);
        float h0 = fsig(o0) * ftanh(c0);
        float h1 = fsig(o1) * ftanh(c1);

        Hs[r0 * HSTR + lane] = (_Float16)h0;
        Hs[r1 * HSTR + lane] = (_Float16)h1;

        // ---- head: a = h . w_lin + b_lin  (wave32 butterfly reduce) ----
        float v0 = h0 * wlinj, v1 = h1 * wlinj;
        #pragma unroll
        for (int off = 16; off > 0; off >>= 1) {
            v0 += __shfl_xor(v0, off, 32);
            v1 += __shfl_xor(v1, off, 32);
        }
        float a0 = v0 + blin, a1 = v1 + blin;
        if (lane == 0)  out[(size_t)(bbase + r0) * TT + t] = a0;
        if (lane == 16) out[(size_t)(bbase + r1) * TT + t] = a1;

        // ---- next step's input: teacher-forced or fed-back prediction ----
        if (t + 1 < T) {
            if (tid < BROWS) Xs[tid] = input[(size_t)(bbase + tid) * T + (t + 1)];
        } else {
            if (lane == 0)  Xs[r0] = a0;
            if (lane == 16) Xs[r1] = a1;
        }
        __syncthreads();
    }
}

extern "C" void kernel_launch(void* const* d_in, const int* in_sizes, int n_in,
                              void* d_out, int out_size, void* d_ws, size_t ws_size,
                              hipStream_t stream) {
    (void)n_in; (void)d_ws; (void)ws_size;
    const float* input = (const float*)d_in[0];
    const float* w_ih  = (const float*)d_in[1];
    const float* w_hh  = (const float*)d_in[2];
    const float* b_ih  = (const float*)d_in[3];
    const float* b_hh  = (const float*)d_in[4];
    const float* w_lin = (const float*)d_in[5];
    const float* b_lin = (const float*)d_in[6];
    float* out = (float*)d_out;

    const int B      = 256;                 // fixed in the reference setup
    const int T      = in_sizes[0] / B;     // input is [B, T, 1]
    const int TT     = out_size / B;        // T + future
    const int future = TT - T;

    lstm_seq_kernel<<<B / BROWS, 256, 0, stream>>>(
        input, w_ih, w_hh, b_ih, b_hh, w_lin, b_lin, out, T, future);
}
